// PoincareTransformerCausal_27685359190112
// MI455X (gfx1250) — compile-verified, matrix operand order checked by reading, not run on hardware
//
#include <hip/hip_runtime.h>
#include <cstdint>

#define BB   2
#define SS   1024
#define DD   512
#define HH   8
#define HDD  64
#define FFD  2048
#define LL   4
#define EPSF 1e-7f
#define MAXN 0.999f
// BETA(32,0.5)/BETA(256,0.5) and its inverse
#define RATIO_SPLIT  2.838119f
#define RATIO_CONCAT 0.3523465f

typedef __bf16 bf16;
typedef __attribute__((ext_vector_type(16))) __bf16 v16bf;
typedef __attribute__((ext_vector_type(8)))  __bf16 v8bf;
typedef __attribute__((ext_vector_type(8)))  float  v8f;
typedef __attribute__((ext_vector_type(4)))  uint32_t u32x4;
typedef __attribute__((ext_vector_type(8)))  uint32_t u32x8;

// ---------------------------------------------------------------- WMMA helpers
__device__ __forceinline__ v8f wmma_bf16(v16bf a, v16bf b, v8f c) {
  return __builtin_amdgcn_wmma_f32_16x16x32_bf16(false, a, false, b, (short)0, c, false, false);
}

// Load a 32-K bf16 fragment (A layout; B uses the mirrored layout with row = n).
// Lane covers K = k0+8h+{0..7} and K = k0+16+8h+{0..7}  (h = lane>>4), row = caller-chosen.
__device__ __forceinline__ v16bf load_frag(const bf16* base, int ld, int row, int k0, int lane) {
  const bf16* p = base + (size_t)row * ld + k0 + ((lane >> 4) << 3);
  v16bf f;
  ((v8bf*)&f)[0] = *(const v8bf*)p;
  ((v8bf*)&f)[1] = *(const v8bf*)(p + 16);
  return f;
}

// ---------------------------------------------------------------- TDM (Tensor Data Mover)
// DMA a [64 rows x 32 cols] bf16 tile (row stride = ld elements) from global into LDS
// (compact row-major). D# per CDNA5 ISA; 2-group form (<=2D tensor).
__device__ __forceinline__ void tdm_load_tile_64x32(uint32_t lds_off, const bf16* gsrc, int ld) {
  uint64_t ga = (uint64_t)(uintptr_t)gsrc;
  u32x4 g0;
  g0[0] = 1u;                                              // count=1, user descriptor
  g0[1] = lds_off;                                         // lds_addr (bytes)
  g0[2] = (uint32_t)ga;                                    // global_addr[95:64]
  g0[3] = ((uint32_t)(ga >> 32) & 0x1FFFFFFu) | (2u << 30); // addr[120:96] | type=2
  u32x8 g1;
  g1[0] = 1u << 16;                                        // wg_mask=0, data_size=1 (2B)
  g1[1] = 0xFFFFu << 16;                                   // tensor_dim0 low16  (huge: no OOB)
  g1[2] = 0x7FFFu | (0xFFFFu << 16);                       // tensor_dim0 hi | tensor_dim1 lo
  g1[3] = 0x7FFFu | (32u << 16);                           // tensor_dim1 hi | tile_dim0=32
  g1[4] = 64u;                                             // tile_dim1=64, tile_dim2=0
  g1[5] = (uint32_t)ld;                                    // tensor_dim0_stride low32
  g1[6] = 0u;                                              // stride hi16 | dim1_stride lo
  g1[7] = 0u;
  asm volatile("tensor_load_to_lds %0, %1" :: "s"(g0), "s"(g1) : "memory");
}

// scale applied to z so that t = scale*z equals logmap0(expmap0(z)) (incl. clipping)
__device__ __forceinline__ float fused_exp_log_scale(float nz) {
  float np  = tanhf(nz);
  float ncl = fminf(fmaxf(np, EPSF), MAXN);
  return atanhf(ncl) / ncl * np / nz;
}

// block-wide sum for 256-thread (8-wave) blocks; sbuf must hold >= 9 floats
__device__ __forceinline__ float block_sum(float v, float* sbuf) {
  __syncthreads();
  for (int off = 16; off; off >>= 1) v += __shfl_xor(v, off);
  if ((threadIdx.x & 31) == 0) sbuf[threadIdx.x >> 5] = v;
  __syncthreads();
  if (threadIdx.x == 0) {
    float t = 0.f;
    int nw = blockDim.x >> 5;
    for (int i = 0; i < nw; ++i) t += sbuf[i];
    sbuf[8] = t;
  }
  __syncthreads();
  return sbuf[8];
}

// ---------------------------------------------------------------- small kernels
__global__ void k_cvt(const float* __restrict__ s, bf16* __restrict__ d, long long n) {
  long long i = (long long)blockIdx.x * blockDim.x + threadIdx.x;
  long long stride = (long long)gridDim.x * blockDim.x;
  for (; i < n; i += stride) d[i] = (bf16)s[i];
}

// out = project(mobius_add(x, y'))  where y' = y (y_tangent=0) or expmap0(y) (y_tangent=1)
__global__ __launch_bounds__(256) void k_mobius(const float* __restrict__ x,
                                                const float* __restrict__ y,
                                                int y_tangent, int y_mod,
                                                float* __restrict__ out, int ncols) {
  __shared__ float sb[16];
  int row = blockIdx.x;
  const float* xr = x + (size_t)row * ncols;
  int yrow = y_mod ? (row % y_mod) : row;
  const float* yr = y + (size_t)yrow * ncols;
  float xv[4], yv[4];
  float sx = 0.f, sy = 0.f, sxy = 0.f;
  int cnt = 0;
  for (int i = threadIdx.x; i < ncols; i += blockDim.x) {
    float a = xr[i], c = yr[i];
    xv[cnt] = a; yv[cnt] = c; ++cnt;
    sx += a * a; sy += c * c; sxy += a * c;
  }
  float us = block_sum(sy, sb);
  float se = 1.f;
  if (y_tangent) { float nu = sqrtf(fmaxf(us, EPSF * EPSF)); se = tanhf(nu) / nu; }
  float x2 = block_sum(sx, sb);
  float xy = se * block_sum(sxy, sb);
  float y2 = se * se * us;
  float c1 = 1.f + 2.f * xy + y2;
  float c2 = 1.f - x2;
  float den = fmaxf(1.f + 2.f * xy + x2 * y2, EPSF);
  float so = 0.f;
  for (int c = 0; c < cnt; ++c) {
    float o = (c1 * xv[c] + c2 * se * yv[c]) / den;
    xv[c] = o; so += o * o;
  }
  float n = sqrtf(fmaxf(block_sum(so, sb), EPSF * EPSF));
  float pf = fminf(1.f, MAXN / n);
  cnt = 0;
  float* orow = out + (size_t)row * ncols;
  for (int i = threadIdx.x; i < ncols; i += blockDim.x) orow[i] = pf * xv[cnt++];
}

// pln(x) followed by the consumer plinear's logmap0, fused; bf16 tangent output
__global__ __launch_bounds__(256) void k_pln(const float* __restrict__ x,
                                             const float* __restrict__ w,
                                             const float* __restrict__ b,
                                             bf16* __restrict__ tout, int ncols) {
  __shared__ float sb[16];
  int row = blockIdx.x;
  const float* xr = x + (size_t)row * ncols;
  float xv[4]; int cnt = 0; float s2 = 0.f;
  for (int i = threadIdx.x; i < ncols; i += blockDim.x) { float v = xr[i]; xv[cnt++] = v; s2 += v * v; }
  float n   = sqrtf(fmaxf(block_sum(s2, sb), EPSF * EPSF));
  float ncl = fminf(fmaxf(n, EPSF), MAXN);
  float sl  = atanhf(ncl) / ncl;                  // logmap0 scale
  float st = 0.f, st2 = 0.f;
  for (int c = 0; c < cnt; ++c) { float t = sl * xv[c]; xv[c] = t; st += t; st2 += t * t; }
  float mu  = block_sum(st, sb) / ncols;
  float m2  = block_sum(st2, sb) / ncols;
  float inv = rsqrtf(fmaxf(m2 - mu * mu, 0.f) + 1e-5f);
  float z2 = 0.f; cnt = 0;
  for (int i = threadIdx.x; i < ncols; i += blockDim.x) {
    float z = (xv[cnt] - mu) * inv * w[i] + b[i];
    xv[cnt] = z; z2 += z * z; ++cnt;
  }
  float nz = sqrtf(fmaxf(block_sum(z2, sb), EPSF * EPSF));
  float f  = fused_exp_log_scale(nz);             // expmap0 then next logmap0
  cnt = 0;
  bf16* tr = tout + (size_t)row * ncols;
  for (int i = threadIdx.x; i < ncols; i += blockDim.x) tr[i] = (bf16)(f * xv[cnt++]);
}

// split_heads of expmap0(u): per-head point (bf16) + squared norm; v-variant -> lam*v (key-transposed) + (lam-1)
__global__ __launch_bounds__(256) void k_split(const float* __restrict__ u,
                                               bf16* __restrict__ hd_out, float* __restrict__ sq_out,
                                               bf16* __restrict__ lamvT, float* __restrict__ lamm1,
                                               int is_v) {
  __shared__ float sb[16];
  int row = blockIdx.x;                        // 0 .. B*S-1
  const float* ur = u + (size_t)row * DD;
  float s2 = 0.f;
  for (int i = threadIdx.x; i < DD; i += 256) { float v = ur[i]; s2 += v * v; }
  float nu = sqrtf(fmaxf(block_sum(s2, sb), EPSF * EPSF));
  float f0 = fused_exp_log_scale(nu) * RATIO_SPLIT;   // expmap0 -> logmap0 -> beta scale
  int head = threadIdx.x >> 5, lane = threadIdx.x & 31;
  float e0 = f0 * ur[head * HDD + lane];
  float e1 = f0 * ur[head * HDD + 32 + lane];
  float nh2 = e0 * e0 + e1 * e1;
  for (int off = 1; off < 32; off <<= 1) nh2 += __shfl_xor(nh2, off);
  float nh = sqrtf(fmaxf(nh2, EPSF * EPSF));
  float th = tanhf(nh);
  float sc = th / nh;                                 // per-head expmap0
  float p0 = sc * e0, p1 = sc * e1;
  int b = row / SS, s = row % SS;
  size_t bh   = (size_t)(b * HH + head);
  size_t sidx = bh * SS + s;
  if (!is_v) {
    hd_out[sidx * HDD + lane]      = (bf16)p0;
    hd_out[sidx * HDD + 32 + lane] = (bf16)p1;
    if (lane == 0) sq_out[sidx] = th * th;
  } else {
    float lam = 2.f / fmaxf(1.f - th * th, EPSF);
    lamvT[(bh * HDD + lane) * SS + s]        = (bf16)(lam * p0);
    lamvT[(bh * HDD + 32 + lane) * SS + s]   = (bf16)(lam * p1);
    if (lane == 0) lamm1[sidx] = lam - 1.f;
  }
}

// fused flash-attention with hyperbolic distance scores + gyromidpoint accumulation
__global__ __launch_bounds__(32) void k_attn(const bf16* __restrict__ qh, const bf16* __restrict__ kh,
                                             const bf16* __restrict__ lamvT,
                                             const float* __restrict__ q2a, const float* __restrict__ k2a,
                                             const float* __restrict__ lamm1a,
                                             const float* __restrict__ taua, const float* __restrict__ gama,
                                             int layer, float* __restrict__ outp) {
  __shared__ float plds[16][32];
  int lane = threadIdx.x;
  int half = lane >> 4, col = lane & 15;
  int bh = blockIdx.y;
  int q0 = blockIdx.x * 16;
  int b = bh / HH, h = bh % HH;
  const bf16* qb  = qh    + (size_t)bh * SS * HDD;
  const bf16* kbm = kh    + (size_t)bh * SS * HDD;
  const bf16* lvb = lamvT + (size_t)bh * HDD * SS;
  const float* q2r = q2a    + (size_t)bh * SS;
  const float* k2r = k2a    + (size_t)bh * SS;
  const float* lm1 = lamm1a + (size_t)bh * SS;
  float tau = taua[layer], gamma = gama[layer];

  v16bf aq0 = load_frag(qb, HDD, q0 + col, 0, lane);
  v16bf aq1 = load_frag(qb, HDD, q0 + col, 32, lane);
  float q2v[8];
  for (int r = 0; r < 8; ++r) q2v[r] = q2r[q0 + r + 8 * half];
  float mrun[8], lrun[8], drun[8];
  for (int r = 0; r < 8; ++r) { mrun[r] = -1e30f; lrun[r] = 0.f; drun[r] = 0.f; }
  v8f accn[4] = {};

  for (int j0 = 0; j0 <= q0 + 15; j0 += 32) {       // 32 keys per step (causal)
    if (j0 + 47 < SS) {                             // prefetch next key block into caches
      __builtin_prefetch(kbm + (size_t)(j0 + 32) * HDD + col * HDD, 0, 1);
    }
    float scf[2][8];
    for (int kb2 = 0; kb2 < 2; ++kb2) {
      int key0 = j0 + 16 * kb2;
      v16bf bk0 = load_frag(kbm, HDD, key0 + col, 0, lane);
      v16bf bk1 = load_frag(kbm, HDD, key0 + col, 32, lane);
      v8f c = {};
      c = wmma_bf16(aq0, bk0, c);
      c = wmma_bf16(aq1, bk1, c);
      float k2c = k2r[key0 + col];
      for (int r = 0; r < 8; ++r) {
        int rowg = q0 + r + 8 * half;
        int colg = key0 + col;
        float qk  = c[r];
        float d2  = fmaxf(q2v[r] + k2c - 2.f * qk, 0.f);
        float den = fmaxf((1.f - q2v[r]) * (1.f - k2c), EPSF);
        float arg = fmaxf(1.f + 2.f * d2 / den, 1.f + 1e-7f);
        float dist = logf(arg + sqrtf(arg * arg - 1.f));  // arccosh
        float s = -tau * dist - gamma;
        if (colg > rowg) s = -1e9f;
        scf[kb2][r] = s;
      }
    }
    float lm1a = lm1[j0 + col], lm1b = lm1[j0 + 16 + col];
    for (int r = 0; r < 8; ++r) {
      float mx = fmaxf(scf[0][r], scf[1][r]);
      for (int off = 1; off < 16; off <<= 1) mx = fmaxf(mx, __shfl_xor(mx, off));
      float mnew  = fmaxf(mrun[r], mx);
      float alpha = __expf(mrun[r] - mnew);
      float p0 = __expf(scf[0][r] - mnew);
      float p1 = __expf(scf[1][r] - mnew);
      float ls = p0 + p1, ds = p0 * lm1a + p1 * lm1b;
      for (int off = 1; off < 16; off <<= 1) { ls += __shfl_xor(ls, off); ds += __shfl_xor(ds, off); }
      lrun[r] = alpha * lrun[r] + ls;
      drun[r] = alpha * drun[r] + ds;
      mrun[r] = mnew;
      for (int t = 0; t < 4; ++t) accn[t][r] *= alpha;
      plds[r + 8 * half][col]      = p0;
      plds[r + 8 * half][16 + col] = p1;
    }
    __syncthreads();
    v16bf pa;                                   // P as A-fragment (row = col, K = local key)
    for (int i = 0; i < 16; ++i) {
      int kk = (i < 8) ? (8 * half + i) : (16 + 8 * half + (i - 8));
      pa[i] = (bf16)plds[col][kk];
    }
    __syncthreads();
    for (int t = 0; t < 4; ++t) {               // P @ (lam*v), d-tiles of 16
      v16bf bv = load_frag(lvb, SS, 16 * t + col, j0, lane);
      accn[t] = wmma_bf16(pa, bv, accn[t]);
    }
  }

  // midpoint + mobius_scalar_mul(0.5) + project; store head-major [B,S,H*64]
  for (int r = 0; r < 8; ++r) {
    float linv = 1.f / lrun[r];
    float denf = fmaxf(drun[r] * linv, EPSF);
    float mid[4], n2 = 0.f;
    for (int t = 0; t < 4; ++t) { mid[t] = accn[t][r] * linv / denf; n2 += mid[t] * mid[t]; }
    for (int off = 1; off < 16; off <<= 1) n2 += __shfl_xor(n2, off);
    float n   = sqrtf(fmaxf(n2, EPSF * EPSF));
    float ncl = fminf(fmaxf(n, EPSF), MAXN);
    float sc2 = tanhf(0.5f * atanhf(ncl)) / ncl;
    float outn = sc2 * n;
    float pf  = fminf(1.f, MAXN / fmaxf(outn, EPSF));
    float f   = sc2 * pf;
    int srow = q0 + r + 8 * half;
    float* orow = outp + ((size_t)b * SS + srow) * DD + h * HDD;
    for (int t = 0; t < 4; ++t) orow[16 * t + col] = f * mid[t];
  }
}

// concat_heads (beta-concat) + Wo-plinear's logmap0, fused -> bf16 tangent
__global__ __launch_bounds__(256) void k_concat(const float* __restrict__ ah, bf16* __restrict__ tout) {
  __shared__ float sb[16];
  int row = blockIdx.x;
  const float* ar = ah + (size_t)row * DD;
  int head = threadIdx.x >> 5, lane = threadIdx.x & 31;
  float a0 = ar[head * HDD + lane], a1 = ar[head * HDD + 32 + lane];
  float n2 = a0 * a0 + a1 * a1;
  for (int off = 1; off < 32; off <<= 1) n2 += __shfl_xor(n2, off);
  float np  = sqrtf(fmaxf(n2, EPSF * EPSF));
  float ncl = fminf(fmaxf(np, EPSF), MAXN);
  float sch = atanhf(ncl) / ncl * RATIO_CONCAT;      // per-head logmap0 + beta scale
  float t0 = sch * a0, t1 = sch * a1;
  float full = block_sum(t0 * t0 + t1 * t1, sb);
  float N = sqrtf(fmaxf(full, EPSF * EPSF));
  float f = fused_exp_log_scale(N);                  // expmap0 then Wo logmap0
  bf16* tr = tout + (size_t)row * DD;
  tr[head * HDD + lane]      = (bf16)(f * t0);
  tr[head * HDD + 32 + lane] = (bf16)(f * t1);
}

// FFN middle: relu in tangent space with both exp/log roundtrips fused
__global__ __launch_bounds__(256) void k_ffn_mid(const float* __restrict__ u, bf16* __restrict__ tout) {
  __shared__ float sb[16];
  int row = blockIdx.x;
  const float* ur = u + (size_t)row * FFD;
  float rv[8]; float s2 = 0.f; int cnt = 0;
  for (int i = threadIdx.x; i < FFD; i += 256) { float v = ur[i]; rv[cnt++] = v; s2 += v * v; }
  float n1 = sqrtf(fmaxf(block_sum(s2, sb), EPSF * EPSF));
  float f1 = fused_exp_log_scale(n1);
  float s3 = 0.f;
  for (int c = 0; c < 8; ++c) { float r = fmaxf(f1 * rv[c], 0.f); rv[c] = r; s3 += r * r; }
  float nr = sqrtf(fmaxf(block_sum(s3, sb), EPSF * EPSF));
  float f2 = fused_exp_log_scale(nr);
  cnt = 0;
  bf16* tr = tout + (size_t)row * FFD;
  for (int i = threadIdx.x; i < FFD; i += 256) tr[i] = (bf16)(f2 * rv[cnt++]);
}

__global__ __launch_bounds__(256) void k_logmap(const float* __restrict__ x, bf16* __restrict__ t, int ncols) {
  __shared__ float sb[16];
  int row = blockIdx.x;
  const float* xr = x + (size_t)row * ncols;
  float xv[4]; int cnt = 0; float s2 = 0.f;
  for (int i = threadIdx.x; i < ncols; i += blockDim.x) { float v = xr[i]; xv[cnt++] = v; s2 += v * v; }
  float n   = sqrtf(fmaxf(block_sum(s2, sb), EPSF * EPSF));
  float ncl = fminf(fmaxf(n, EPSF), MAXN);
  float f   = atanhf(ncl) / ncl;
  cnt = 0;
  bf16* tr = t + (size_t)row * ncols;
  for (int i = threadIdx.x; i < ncols; i += blockDim.x) tr[i] = (bf16)(f * xv[cnt++]);
}

__global__ __launch_bounds__(256) void k_expmap(const float* __restrict__ u, float* __restrict__ o, int ncols) {
  __shared__ float sb[16];
  int row = blockIdx.x;
  const float* ur = u + (size_t)row * ncols;
  float uv[4]; int cnt = 0; float s2 = 0.f;
  for (int i = threadIdx.x; i < ncols; i += blockDim.x) { float v = ur[i]; uv[cnt++] = v; s2 += v * v; }
  float n = sqrtf(fmaxf(block_sum(s2, sb), EPSF * EPSF));
  float f = tanhf(n) / n;
  cnt = 0;
  float* orow = o + (size_t)row * ncols;
  for (int i = threadIdx.x; i < ncols; i += blockDim.x) orow[i] = f * uv[cnt++];
}

// C[M,N] = A[M,K](bf16) @ W[N,K]^T(bf16) + bias
// 4-wave workgroup, 64x64 tile; A/B 64x32 tiles staged global->LDS by TDM, double-buffered.
__global__ __launch_bounds__(128) void k_gemm(const bf16* __restrict__ A, const bf16* __restrict__ W,
                                              const float* __restrict__ bias, float* __restrict__ C,
                                              int M, int N, int K) {
  __shared__ bf16 lA[2][64 * 32];
  __shared__ bf16 lB[2][64 * 32];
  (void)M;
  int tid = threadIdx.x;
  int wave = tid >> 5, lane = tid & 31;
  int mblk = blockIdx.y * 64;
  int n0   = blockIdx.x * 64;
  v8f acc[4] = {};
  if (wave == 0) {
    tdm_load_tile_64x32((uint32_t)(uintptr_t)&lA[0][0], A + (size_t)mblk * K, K);
    tdm_load_tile_64x32((uint32_t)(uintptr_t)&lB[0][0], W + (size_t)n0 * K, K);
  }
  for (int k0 = 0; k0 < K; k0 += 32) {
    int buf = (k0 >> 5) & 1;
    bool more = (k0 + 32) < K;
    if (wave == 0) {
      if (more) {
        tdm_load_tile_64x32((uint32_t)(uintptr_t)&lA[buf ^ 1][0], A + (size_t)mblk * K + k0 + 32, K);
        tdm_load_tile_64x32((uint32_t)(uintptr_t)&lB[buf ^ 1][0], W + (size_t)n0 * K + k0 + 32, K);
        __builtin_amdgcn_s_wait_tensorcnt(2);   // current buffer's 2 DMAs done (in-order)
      } else {
        __builtin_amdgcn_s_wait_tensorcnt(0);
      }
    }
    __syncthreads();                            // publish LDS tiles to all 4 waves
    v16bf a = load_frag(&lA[buf][0], 32, wave * 16 + (lane & 15), 0, lane);
    for (int t = 0; t < 4; ++t) {
      v16bf b = load_frag(&lB[buf][0], 32, 16 * t + (lane & 15), 0, lane);
      acc[t] = wmma_bf16(a, b, acc[t]);
    }
    __syncthreads();                            // tiles consumed before next overwrite
  }
  int half = lane >> 4, col = lane & 15;
  int m0 = mblk + wave * 16;
  for (int t = 0; t < 4; ++t) {
    int n = n0 + 16 * t + col;
    float bi = bias[n];
    for (int r = 0; r < 8; ++r) {
      int m = m0 + r + 8 * half;
      C[(size_t)m * N + n] = acc[t][r] + bi;
    }
  }
}

// ---------------------------------------------------------------- launch
extern "C" void kernel_launch(void* const* d_in, const int* in_sizes, int n_in,
                              void* d_out, int out_size, void* d_ws, size_t ws_size,
                              hipStream_t stream) {
  (void)in_sizes; (void)n_in; (void)out_size; (void)ws_size;
  const float* x_in = (const float*)d_in[0];
  const float* pos  = (const float*)d_in[1];
  const float* ln1w = (const float*)d_in[2];
  const float* ln1b = (const float*)d_in[3];
  const float* ln2w = (const float*)d_in[4];
  const float* ln2b = (const float*)d_in[5];
  const float* Wq   = (const float*)d_in[6];
  const float* bq   = (const float*)d_in[7];
  const float* Wk   = (const float*)d_in[8];
  const float* bk   = (const float*)d_in[9];
  const float* Wv   = (const float*)d_in[10];
  const float* bvp  = (const float*)d_in[11];
  const float* Wo   = (const float*)d_in[12];
  const float* bo   = (const float*)d_in[13];
  const float* W1   = (const float*)d_in[14];
  const float* b1   = (const float*)d_in[15];
  const float* W2   = (const float*)d_in[16];
  const float* b2   = (const float*)d_in[17];
  const float* tau  = (const float*)d_in[18];
  const float* gam  = (const float*)d_in[19];
  const float* Wout = (const float*)d_in[20];
  const float* bout = (const float*)d_in[21];

  char* base = (char*)d_ws;
  size_t off = 0;
  auto alloc = [&](size_t bytes) -> void* {
    void* r = base + off;
    off += (bytes + 255) & ~(size_t)255;
    return r;
  };
  const size_t R   = (size_t)BB * SS;       // 2048 rows
  const size_t BHS = (size_t)BB * HH * SS;  // 16384
  float* X    = (float*)alloc(R * DD * 4);
  float* U    = (float*)alloc(R * FFD * 4);
  float* AH   = (float*)alloc(R * DD * 4);
  float* Q2   = (float*)alloc(BHS * 4);
  float* K2   = (float*)alloc(BHS * 4);
  float* LM1  = (float*)alloc(BHS * 4);
  bf16* TB    = (bf16*)alloc(R * FFD * 2);
  bf16* QH    = (bf16*)alloc(BHS * HDD * 2);
  bf16* KH    = (bf16*)alloc(BHS * HDD * 2);
  bf16* LVT   = (bf16*)alloc(BHS * HDD * 2);
  bf16* WQB   = (bf16*)alloc((size_t)LL * DD * DD * 2);
  bf16* WKB   = (bf16*)alloc((size_t)LL * DD * DD * 2);
  bf16* WVB   = (bf16*)alloc((size_t)LL * DD * DD * 2);
  bf16* WOB   = (bf16*)alloc((size_t)LL * DD * DD * 2);
  bf16* W1B   = (bf16*)alloc((size_t)LL * FFD * DD * 2);
  bf16* W2B   = (bf16*)alloc((size_t)LL * DD * FFD * 2);
  bf16* WOUTB = (bf16*)alloc((size_t)DD * DD * 2);

  auto cvt = [&](const float* s, bf16* d, size_t n) {
    k_cvt<<<dim3((unsigned)((n + 255) / 256)), dim3(256), 0, stream>>>(s, d, (long long)n);
  };
  cvt(Wq, WQB, (size_t)LL * DD * DD);
  cvt(Wk, WKB, (size_t)LL * DD * DD);
  cvt(Wv, WVB, (size_t)LL * DD * DD);
  cvt(Wo, WOB, (size_t)LL * DD * DD);
  cvt(W1, W1B, (size_t)LL * FFD * DD);
  cvt(W2, W2B, (size_t)LL * DD * FFD);
  cvt(Wout, WOUTB, (size_t)DD * DD);

  dim3 rb((unsigned)R), tb(256);
  dim3 gD(DD / 64, (unsigned)(R / 64));
  dim3 gF(FFD / 64, (unsigned)(R / 64));

  // x = mobius_add(x_in, pos_tab)
  k_mobius<<<rb, tb, 0, stream>>>(x_in, pos, 0, SS, X, DD);

  for (int l = 0; l < LL; ++l) {
    const bf16* wq = WQB + (size_t)l * DD * DD;
    const bf16* wk = WKB + (size_t)l * DD * DD;
    const bf16* wv = WVB + (size_t)l * DD * DD;
    const bf16* wo = WOB + (size_t)l * DD * DD;
    const bf16* w1 = W1B + (size_t)l * FFD * DD;
    const bf16* w2 = W2B + (size_t)l * DD * FFD;

    // ---- attention block
    k_pln<<<rb, tb, 0, stream>>>(X, ln1w + l * DD, ln1b + l * DD, TB, DD);
    k_gemm<<<gD, 128, 0, stream>>>(TB, wq, bq + l * DD, U, (int)R, DD, DD);
    k_split<<<rb, tb, 0, stream>>>(U, QH, Q2, LVT, LM1, 0);
    k_gemm<<<gD, 128, 0, stream>>>(TB, wk, bk + l * DD, U, (int)R, DD, DD);
    k_split<<<rb, tb, 0, stream>>>(U, KH, K2, LVT, LM1, 0);
    k_gemm<<<gD, 128, 0, stream>>>(TB, wv, bvp + l * DD, U, (int)R, DD, DD);
    k_split<<<rb, tb, 0, stream>>>(U, QH, Q2, LVT, LM1, 1);   // writes LVT/LM1 only
    k_attn<<<dim3(SS / 16, BB * HH), 32, 0, stream>>>(QH, KH, LVT, Q2, K2, LM1, tau, gam, l, AH);
    k_concat<<<rb, tb, 0, stream>>>(AH, TB);
    k_gemm<<<gD, 128, 0, stream>>>(TB, wo, bo + l * DD, U, (int)R, DD, DD);
    k_mobius<<<rb, tb, 0, stream>>>(X, U, 1, 0, X, DD);

    // ---- FFN block
    k_pln<<<rb, tb, 0, stream>>>(X, ln2w + l * DD, ln2b + l * DD, TB, DD);
    k_gemm<<<gF, 128, 0, stream>>>(TB, w1, b1 + l * FFD, U, (int)R, FFD, DD);
    k_ffn_mid<<<rb, tb, 0, stream>>>(U, TB);
    k_gemm<<<gD, 128, 0, stream>>>(TB, w2, b2 + l * DD, U, (int)R, DD, FFD);
    k_mobius<<<rb, tb, 0, stream>>>(X, U, 1, 0, X, DD);
  }

  // final plinear(x, Wout, bout)
  k_logmap<<<rb, tb, 0, stream>>>(X, TB, DD);
  k_gemm<<<gD, 128, 0, stream>>>(TB, WOUTB, bout, U, (int)R, DD, DD);
  k_expmap<<<rb, tb, 0, stream>>>(U, (float*)d_out, DD);
}
